// DiffusionModel_89910845375049
// MI455X (gfx1250) — matrix operand order repaired and checked by reading, not compiled, not run on hardware
//
#include <hip/hip_runtime.h>
#include <hip/hip_bf16.h>
#include <math.h>

// ---------------------------------------------------------------------------
// GATv2 diffusion GNN forward for MI455X (gfx1250), wave32 + WMMA.
//   N=20000 nodes, E=640000 edges, HID=128, H=4, C=128 (H*C=512), ED=4.
// Dense transforms: v_wmma_f32_16x16x32_f16, 16x64 tile per wave, K unrolled
// at compile time, dual-weight fusion for the GAT xl/xr transforms.
// Edge softmax/aggregation: L2-resident gathers + f32 global atomics.
// ---------------------------------------------------------------------------

#define NN   20000
#define NE   640000
#define HID  128
#define HC   512
#define KPAD 224   // 2 + 64 + 0 + 128 = 194 -> padded to multiple of 32

typedef __attribute__((ext_vector_type(16))) _Float16 v16h;
typedef __attribute__((ext_vector_type(8)))  float    v8f;

__device__ __forceinline__ float gelu_f(float v) {
    return 0.5f * v * (1.f + erff(v * 0.70710678118654752f));
}

// ---------------------------------------------------------------------------
// WMMA GEMM: C[N,M] = act(A[N,K] @ W[M,K]^T + bias)
// One wave computes a 16x64 output tile: one A fragment feeds 4 (or 8, if
// DUAL) v_wmma_f32_16x16x32_f16 per 32-wide K chunk. K is a template const
// so the chunk loop fully unrolls. DUAL computes a second GEMM with the same
// A (used for the GAT xl/xr pair, halving A traffic).
// grid = (M/64, Nrows/16), block = 32.
// A/B fragments use contiguous k = 16*(lane>>4) + elem; f32 C/D layout per CDNA5 ISA 7.12.2.
// ACT: 0=none, 1=relu, 2=gelu (applied to C0 only; DUAL outputs use ACT=0).
// ---------------------------------------------------------------------------
template<int K, int ACT, bool DUAL>
__global__ __launch_bounds__(32) void k_gemm(
    const float* __restrict__ A, int lda,
    const float* __restrict__ W0, const float* __restrict__ bias0, float* __restrict__ C0,
    const float* __restrict__ W1, const float* __restrict__ bias1, float* __restrict__ C1,
    int ldw, int ldc)
{
    const int lane    = threadIdx.x;          // 0..31
    const int colBase = blockIdx.x * 64;      // output-feature tile (4x16)
    const int rowBase = blockIdx.y * 16;      // node-row tile
    const int m  = lane & 15;
    const int hi = lane >> 4;                 // 0 | 1

    const float* arow = A + (size_t)(rowBase + m) * lda;

    v8f acc0[4] = {};
    v8f acc1[4] = {};

    #pragma unroll
    for (int k0 = 0; k0 < K; k0 += 32) {
        // --- A fragment (16x32 f16): vgpr i<4: K=2i|2i+1 (+16 for hi half);
        //     vgpr i>=4: K=8+2(i-4) (+16 for hi half)
        v16h a;
        #pragma unroll
        for (int i = 0; i < 4; ++i) {
            float2 f = *(const float2*)(arow + k0 + 2 * i + (hi ? 16 : 0));
            a[2 * i]     = (_Float16)f.x;
            a[2 * i + 1] = (_Float16)f.y;
        }
        #pragma unroll
        for (int i = 4; i < 8; ++i) {
            float2 f = *(const float2*)(arow + k0 + 8 + 2 * (i - 4) + (hi ? 16 : 0));
            a[2 * i]     = (_Float16)f.x;
            a[2 * i + 1] = (_Float16)f.y;
        }
        // --- 4 column tiles share the A fragment
        #pragma unroll
        for (int t = 0; t < 4; ++t) {
            // B fragment (32x16 f16): lanes 0-15 K=0..15, lanes 16-31 K=16..31
            const float* w0row = W0 + (size_t)(colBase + t * 16 + m) * ldw;
            v16h b0;
            #pragma unroll
            for (int v = 0; v < 8; ++v) {
                float2 f = *(const float2*)(w0row + k0 + 2 * v + (hi ? 16 : 0));
                b0[2 * v]     = (_Float16)f.x;
                b0[2 * v + 1] = (_Float16)f.y;
            }
            acc0[t] = __builtin_amdgcn_wmma_f32_16x16x32_f16(
                false, a, false, b0, (short)0, acc0[t], false, false);
            if constexpr (DUAL) {
                const float* w1row = W1 + (size_t)(colBase + t * 16 + m) * ldw;
                v16h b1;
                #pragma unroll
                for (int v = 0; v < 8; ++v) {
                    float2 f = *(const float2*)(w1row + k0 + 2 * v + (hi ? 16 : 0));
                    b1[2 * v]     = (_Float16)f.x;
                    b1[2 * v + 1] = (_Float16)f.y;
                }
                acc1[t] = __builtin_amdgcn_wmma_f32_16x16x32_f16(
                    false, a, false, b1, (short)0, acc1[t], false, false);
            }
        }
    }

    // --- epilogue: D layout is N=lane&15 (column), M=vgpr + 8*(lane>=16)
    const int n = lane & 15;
    #pragma unroll
    for (int t = 0; t < 4; ++t) {
        const int col = colBase + t * 16 + n;
        const float bv0 = bias0 ? bias0[col] : 0.f;
        const float bv1 = (DUAL && bias1) ? bias1[col] : 0.f;
        #pragma unroll
        for (int r = 0; r < 8; ++r) {
            int row = rowBase + r + (hi ? 8 : 0);
            float v = acc0[t][r] + bv0;
            if (ACT == 1)      v = fmaxf(v, 0.f);
            else if (ACT == 2) v = gelu_f(v);
            C0[(size_t)row * ldc + col] = v;
            if constexpr (DUAL)
                C1[(size_t)row * ldc + col] = acc1[t][r] + bv1;
        }
    }
}

// ---------------------------------------------------------------------------
// Time embedding: te = gelu(time_w * t + time_b), [128]
// ---------------------------------------------------------------------------
__global__ void k_time(const float* __restrict__ t,
                       const float* __restrict__ tw,
                       const float* __restrict__ tb,
                       float* __restrict__ te)
{
    int i = threadIdx.x;
    te[i] = gelu_f(tw[i] * t[0] + tb[i]);
}

// ---------------------------------------------------------------------------
// Pad in1_w [128,194] -> [128,224] with zeros.
// ---------------------------------------------------------------------------
__global__ void k_padw(const float* __restrict__ w, float* __restrict__ wp)
{
    int t = blockIdx.x * blockDim.x + threadIdx.x;
    if (t >= 128 * KPAD) return;
    int r = t / KPAD, c = t % KPAD;
    wp[t] = (c < 194) ? w[r * 194 + c] : 0.f;
}

// ---------------------------------------------------------------------------
// Build h0 [N,224] = [gelu(xenc), posenc(64), te(128), zero-pad(30)]
// ---------------------------------------------------------------------------
__global__ void k_embed(const float* __restrict__ x,
                        const float* __restrict__ xw,
                        const float* __restrict__ xb,
                        const float* __restrict__ te,
                        float* __restrict__ h0)
{
    long t = (long)blockIdx.x * blockDim.x + threadIdx.x;
    if (t >= (long)NN * KPAD) return;
    int n = (int)(t / KPAD), col = (int)(t % KPAD);
    float val = 0.f;
    if (col < 2) {
        float v = xw[col * 2 + 0] * x[n * 2 + 0] + xw[col * 2 + 1] * x[n * 2 + 1] + xb[col];
        val = gelu_f(v);
    } else if (col < 66) {
        int k = col - 2;                 // 0..63 ; [0,32)=x0, [32,64)=x1
        float coord = x[n * 2 + (k >> 5)];
        int w = k & 31;                  // within-axis index, pairs (sin,cos)
        float div = __expf(-(float)(w & ~1) * (logf(10000.f) / 32.f));
        float a = coord * div;
        val = (w & 1) ? __cosf(a) : __sinf(a);
    } else if (col < 194) {
        val = te[col - 66];
    }
    h0[(size_t)n * KPAD + col] = val;
}

// monotonic float<->uint mapping for atomicMax on floats
__device__ __forceinline__ unsigned f2ord(float f) {
    unsigned u = __float_as_uint(f);
    return (u & 0x80000000u) ? ~u : (u | 0x80000000u);
}
__device__ __forceinline__ float ord2f(unsigned u) {
    return (u & 0x80000000u) ? __uint_as_float(u ^ 0x80000000u) : __uint_as_float(~u);
}

// ---------------------------------------------------------------------------
// Pass 1: per-edge attention logits + segment-max into dst.
// One wave per edge; lanes split the 128 channels per head.
// ---------------------------------------------------------------------------
__global__ __launch_bounds__(256) void k_edge_logits(
    const int* __restrict__ ei, const float* __restrict__ eattr,
    const float* __restrict__ xl, const float* __restrict__ xr,
    const float* __restrict__ We, const float* __restrict__ att,
    float* __restrict__ logit, unsigned* __restrict__ mx)
{
    int e    = (blockIdx.x * blockDim.x + threadIdx.x) >> 5;
    int lane = threadIdx.x & 31;
    if (e >= NE) return;
    int src = ei[e];
    int dst = ei[NE + e];
    float ea0 = eattr[e * 4 + 0], ea1 = eattr[e * 4 + 1];
    float ea2 = eattr[e * 4 + 2], ea3 = eattr[e * 4 + 3];
    const float* xls = xl + (size_t)src * HC;
    const float* xrd = xr + (size_t)dst * HC;
    #pragma unroll
    for (int h = 0; h < 4; ++h) {
        float s = 0.f;
        #pragma unroll
        for (int j = 0; j < 4; ++j) {
            int c   = lane + 32 * j;
            int idx = h * 128 + c;
            float ev = ea0 * We[idx * 4 + 0] + ea1 * We[idx * 4 + 1]
                     + ea2 * We[idx * 4 + 2] + ea3 * We[idx * 4 + 3];
            float m = xls[idx] + xrd[idx] + ev;
            m = (m > 0.f) ? m : 0.2f * m;        // leaky_relu(0.2)
            s += att[idx] * m;
        }
        for (int off = 16; off > 0; off >>= 1) s += __shfl_down(s, off, 32);
        if (lane == 0) {
            logit[(size_t)e * 4 + h] = s;
            atomicMax(&mx[dst * 4 + h], f2ord(s));
        }
    }
}

// ---------------------------------------------------------------------------
// Pass 2: ex = exp(logit - mx[dst]); den[dst] += ex; logit <- ex.
// ---------------------------------------------------------------------------
__global__ void k_edge_softmax(const int* __restrict__ ei,
                               float* __restrict__ logit,
                               const unsigned* __restrict__ mx,
                               float* __restrict__ den)
{
    int t = blockIdx.x * blockDim.x + threadIdx.x;
    if (t >= NE * 4) return;
    int e = t >> 2, h = t & 3;
    int dst = ei[NE + e];
    float ex = __expf(logit[t] - ord2f(mx[dst * 4 + h]));
    logit[t] = ex;
    atomicAdd(&den[dst * 4 + h], ex);
}

// ---------------------------------------------------------------------------
// Pass 3: agg[dst] += alpha * xl[src].  One wave per edge, 16 elems/lane.
// ---------------------------------------------------------------------------
__global__ __launch_bounds__(256) void k_edge_aggregate(
    const int* __restrict__ ei, const float* __restrict__ logit,
    const float* __restrict__ den, const float* __restrict__ xl,
    float* __restrict__ agg)
{
    int e    = (blockIdx.x * blockDim.x + threadIdx.x) >> 5;
    int lane = threadIdx.x & 31;
    if (e >= NE) return;
    int src = ei[e], dst = ei[NE + e];
    float alpha[4];
    #pragma unroll
    for (int h = 0; h < 4; ++h)
        alpha[h] = logit[(size_t)e * 4 + h] / (den[dst * 4 + h] + 1e-16f);
    const float* xls = xl + (size_t)src * HC;
    float* ag = agg + (size_t)dst * HC;
    #pragma unroll
    for (int j = 0; j < 16; ++j) {
        int idx = lane + 32 * j;           // idx>>7 == j>>2 (constant per j)
        atomicAdd(&ag[idx], alpha[j >> 2] * xls[idx]);
    }
}

// ---------------------------------------------------------------------------
// Node epilogue (in place on h): v = f(src) [+ h] + te ; optional LN+relu.
//   aggMode=1: f(src) = mean over 4 heads of agg + gat_bias
//   aggMode=0: f(src) = src[n*HID + c]  (mlp output, bias already applied)
// One wave per node, 4 features per lane. doLN -> layernorm + relu.
// ---------------------------------------------------------------------------
__global__ __launch_bounds__(128) void k_node_combine(
    float* __restrict__ h, const float* __restrict__ srcA, int aggMode,
    const float* __restrict__ gbias, int useResid,
    const float* __restrict__ te,
    const float* __restrict__ lng, const float* __restrict__ lnb, int doLN)
{
    int n    = (blockIdx.x * blockDim.x + threadIdx.x) >> 5;
    int lane = threadIdx.x & 31;
    if (n >= NN) return;
    float v[4];
    #pragma unroll
    for (int j = 0; j < 4; ++j) {
        int c = lane + 32 * j;
        float a;
        if (aggMode) {
            const float* s = srcA + (size_t)n * HC;
            a = 0.25f * (s[c] + s[128 + c] + s[256 + c] + s[384 + c]) + gbias[c];
        } else {
            a = srcA[(size_t)n * HID + c];
        }
        if (useResid) a += h[(size_t)n * HID + c];
        v[j] = a + te[c];
    }
    if (doLN) {
        float s = v[0] + v[1] + v[2] + v[3];
        for (int off = 16; off > 0; off >>= 1) s += __shfl_xor(s, off, 32);
        float mean = s * (1.f / 128.f);
        float q = 0.f;
        #pragma unroll
        for (int j = 0; j < 4; ++j) { float d = v[j] - mean; q += d * d; }
        for (int off = 16; off > 0; off >>= 1) q += __shfl_xor(q, off, 32);
        float rstd = rsqrtf(q * (1.f / 128.f) + 1e-5f);
        #pragma unroll
        for (int j = 0; j < 4; ++j) {
            int c = lane + 32 * j;
            v[j] = fmaxf((v[j] - mean) * rstd * lng[c] + lnb[c], 0.f);
        }
    }
    #pragma unroll
    for (int j = 0; j < 4; ++j)
        h[(size_t)n * HID + lane + 32 * j] = v[j];
}

// ---------------------------------------------------------------------------
// Final projection: out[n,2] = h[n] @ out_w^T + out_b
// ---------------------------------------------------------------------------
__global__ void k_out(const float* __restrict__ h, const float* __restrict__ ow,
                      const float* __restrict__ ob, float* __restrict__ out)
{
    int t = blockIdx.x * blockDim.x + threadIdx.x;
    if (t >= NN * 2) return;
    int n = t >> 1, j = t & 1;
    const float* hr = h + (size_t)n * HID;
    const float* wr = ow + j * HID;
    float s = ob[j];
    #pragma unroll 4
    for (int k = 0; k < HID; ++k) s += hr[k] * wr[k];
    out[t] = s;
}

// ---------------------------------------------------------------------------
// Host side. d_in order: x, edge_index, edge_attr, t, node_attr, then params
// leaves in jax tree-flatten (sorted-key) order.
// ---------------------------------------------------------------------------
extern "C" void kernel_launch(void* const* d_in, const int* in_sizes, int n_in,
                              void* d_out, int out_size, void* d_ws, size_t ws_size,
                              hipStream_t stream)
{
    (void)in_sizes; (void)n_in; (void)out_size; (void)ws_size;
    const float* x     = (const float*)d_in[0];
    const int*   ei    = (const int*)d_in[1];
    const float* eattr = (const float*)d_in[2];
    const float* tval  = (const float*)d_in[3];
    // d_in[4] = node_attr [N,0] (empty)

    auto Pf = [&](int i) { return (const float*)d_in[5 + i]; };
    // Per block (46 leaves): gnnG at base+G*7 {We,Wl,Wr,att,bias,bl,br},
    // lnL at base+28+L*2 {g,b}, mlpM at base+38+M*4 {b1,b2,w1,w2}.
    const int iIn1b = 92, iIn1w = 93, iIn2b = 94, iIn2w = 95;
    const int iOutb = 96, iOutw = 97, iTimeb = 98, iTimew = 99;
    const int iXb = 100, iXw = 101;

    // ---- workspace carve-up (floats), ~155 MB total ----
    float* ws = (float*)d_ws;
    size_t o = 0;
    auto alloc = [&](size_t nf) { float* p = ws + o; o += (nf + 63) & ~(size_t)63; return p; };
    float* teb   = alloc(128);
    float* hA    = alloc((size_t)NN * HID);
    float* hB    = alloc((size_t)NN * HID);
    float* xl    = alloc((size_t)NN * HC);
    float* xr    = alloc((size_t)NN * HC);
    float* agg   = alloc((size_t)NN * HC);
    float* logit = alloc((size_t)NE * 4);
    float* w1pad = alloc((size_t)128 * KPAD);
    float* mxf   = alloc((size_t)NN * 4);   // used as unsigned
    float* den   = alloc((size_t)NN * 4);
    float* h0    = agg;  // alias: h0 [N,224] only live before first GAT memset

    // ---- input stage ----
    k_time<<<1, 128, 0, stream>>>(tval, Pf(iTimew), Pf(iTimeb), teb);
    k_padw<<<(128 * KPAD + 255) / 256, 256, 0, stream>>>(Pf(iIn1w), w1pad);
    k_embed<<<(int)(((size_t)NN * KPAD + 255) / 256), 256, 0, stream>>>(
        x, Pf(iXw), Pf(iXb), teb, h0);

    const dim3 gNarrow(HID / 64, NN / 16);  // M=128
    const dim3 gWide(HC / 64, NN / 16);     // M=512

    k_gemm<KPAD, 2, false><<<gNarrow, 32, 0, stream>>>(
        h0, KPAD, w1pad, Pf(iIn1b), hA, nullptr, nullptr, nullptr, KPAD, HID);
    k_gemm<HID, 2, false><<<gNarrow, 32, 0, stream>>>(
        hA, HID, Pf(iIn2w), Pf(iIn2b), hB, nullptr, nullptr, nullptr, HID, HID);
    float* h = hB;

    const int edgeWaveGrid = (NE * 32 + 255) / 256;
    const int nodeWaveGrid = (NN * 32 + 127) / 128;

    for (int b = 0; b < 2; ++b) {
        const int base = b * 46;

        auto runGat = [&](int g, int lnIdx, int useResid) {
            const float* We  = Pf(base + g * 7 + 0);
            const float* Wl  = Pf(base + g * 7 + 1);
            const float* Wr  = Pf(base + g * 7 + 2);
            const float* att = Pf(base + g * 7 + 3);
            const float* gb  = Pf(base + g * 7 + 4);
            const float* bl  = Pf(base + g * 7 + 5);
            const float* br  = Pf(base + g * 7 + 6);
            // fused xl/xr: one A read feeds both weight sets
            k_gemm<HID, 0, true><<<gWide, 32, 0, stream>>>(
                h, HID, Wl, bl, xl, Wr, br, xr, HID, HC);
            hipMemsetAsync(mxf, 0, (size_t)NN * 4 * sizeof(float), stream);
            hipMemsetAsync(den, 0, (size_t)NN * 4 * sizeof(float), stream);
            hipMemsetAsync(agg, 0, (size_t)NN * HC * sizeof(float), stream);
            k_edge_logits<<<edgeWaveGrid, 256, 0, stream>>>(
                ei, eattr, xl, xr, We, att, logit, (unsigned*)mxf);
            k_edge_softmax<<<(NE * 4 + 255) / 256, 256, 0, stream>>>(
                ei, logit, (const unsigned*)mxf, den);
            k_edge_aggregate<<<edgeWaveGrid, 256, 0, stream>>>(ei, logit, den, xl, agg);
            k_node_combine<<<nodeWaveGrid, 128, 0, stream>>>(
                h, agg, 1, gb, useResid, teb,
                Pf(base + 28 + lnIdx * 2), Pf(base + 28 + lnIdx * 2 + 1), 1);
        };

        auto runMlp = [&](int m, int lnIdx /* <0: none */) {
            const float* b1 = Pf(base + 38 + m * 4 + 0);
            const float* b2 = Pf(base + 38 + m * 4 + 1);
            const float* w1 = Pf(base + 38 + m * 4 + 2);
            const float* w2 = Pf(base + 38 + m * 4 + 3);
            k_gemm<HID, 1, false><<<gWide, 32, 0, stream>>>(
                h, HID, w1, b1, xl, nullptr, nullptr, nullptr, HID, HC);
            k_gemm<HC, 0, false><<<gNarrow, 32, 0, stream>>>(
                xl, HC, w2, b2, xr, nullptr, nullptr, nullptr, HC, HID);
            if (lnIdx < 0)   // h = h + mlp(h) + te
                k_node_combine<<<nodeWaveGrid, 128, 0, stream>>>(
                    h, xr, 0, nullptr, 1, teb, nullptr, nullptr, 0);
            else             // h = relu(ln(mlp(h) + te))
                k_node_combine<<<nodeWaveGrid, 128, 0, stream>>>(
                    h, xr, 0, nullptr, 0, teb,
                    Pf(base + 28 + lnIdx * 2), Pf(base + 28 + lnIdx * 2 + 1), 1);
        };

        runGat(0, 0, 1);   // h = relu(ln1(gat1(h) + h + te))
        runGat(1, 1, 1);   // h = relu(ln2(gat2(h) + h + te))
        runMlp(0, -1);     // h = h + mlp1(h) + te
        runGat(2, 2, 0);   // h = relu(ln3(gat3(h) + te))
        runGat(3, 3, 0);   // h = relu(ln4(gat4(h) + te))
        runMlp(1, 4);      // h = relu(ln5(mlp2(h) + te))
    }

    k_out<<<(NN * 2 + 255) / 256, 256, 0, stream>>>(h, Pf(iOutw), Pf(iOutb), (float*)d_out);
}
